// SwitchMoEWrapper_7834020348509
// MI455X (gfx1250) — matrix operand order repaired and checked by reading, not compiled, not run on hardware
//
#include <hip/hip_runtime.h>
#include <hip/hip_bf16.h>

// Problem constants (match reference)
#define BB 2
#define SS 1024
#define HH 768
#define FF 3072
#define EE 8
#define BS (BB * SS)          // 2048 tokens
#define TM 32                 // token tile (M) per block: two 16-row WMMA tiles
#define TILES (BS / TM)       // 64 max tiles per expert
#define FCH (FF / 128)        // 24 f-chunks of 128
#define KCH (HH / 32)         // 24 k-chunks of 32 (GEMM1)
#define K2CH (FF / 32)        // 96 k-chunks of 32 (GEMM2)
#define W1_ITEMS 2            // per-thread async b128 issues per W1 chunk
#define W2_ITEMS 12           // per-thread async b128 issues per W2 chunk

typedef __attribute__((ext_vector_type(8)))  float  v8f;
typedef __attribute__((ext_vector_type(4)))  __bf16 v4bf;
typedef __attribute__((ext_vector_type(8)))  __bf16 v8bf;
typedef __attribute__((ext_vector_type(16))) __bf16 v16bf;

#define CAT16(lo, hi) __builtin_shufflevector((lo), (hi), 0, 1, 2, 3, 4, 5, 6, 7, \
                                              8, 9, 10, 11, 12, 13, 14, 15)

// --------------------------- async copy to LDS ------------------------------
#if defined(__HIP_DEVICE_COMPILE__) && \
    __has_builtin(__builtin_amdgcn_global_load_async_to_lds_b128)
#define HAVE_ASYNC_LDS 1
typedef __attribute__((vector_size(16))) int v4i_vs;
typedef __attribute__((address_space(1))) v4i_vs* gv4i_t;   // global int4*
typedef __attribute__((address_space(3))) v4i_vs* lv4i_t;   // LDS int4*
__device__ __forceinline__ void async_cp16(const __bf16* g, __bf16* l) {
    void* gv = (void*)g;
    void* lv = (void*)l;
    __builtin_amdgcn_global_load_async_to_lds_b128((gv4i_t)gv, (lv4i_t)lv, 0, 0);
}
template <int N> __device__ __forceinline__ void async_waitN() {
#if __has_builtin(__builtin_amdgcn_s_wait_asynccnt)
    __builtin_amdgcn_s_wait_asynccnt(N);
#else
    asm volatile("s_wait_asynccnt %0" ::"i"(N) : "memory");
#endif
}
#else
__device__ __forceinline__ void async_cp16(const __bf16* g, __bf16* l) {
    *(v8bf*)l = *(const v8bf*)g;          // sync fallback: b128 load + b128 ds store
}
template <int N> __device__ __forceinline__ void async_waitN() {}
#endif

// ---------------------------------------------------------------------------
// Kernel 0: zero the per-expert token counters
// ---------------------------------------------------------------------------
__global__ void moe_zero_counts(int* __restrict__ counts) {
    if (threadIdx.x < EE) counts[threadIdx.x] = 0;
}

// ---------------------------------------------------------------------------
// Kernel 1: router.  One thread per token.
// ---------------------------------------------------------------------------
__global__ __launch_bounds__(256) void moe_router(
    const float* __restrict__ x, const float* __restrict__ Wr,
    float* __restrict__ logits_out, int* __restrict__ eidx_out,
    float* __restrict__ prob_ws, int* __restrict__ counts,
    int* __restrict__ tlist)
{
    int t = blockIdx.x * blockDim.x + threadIdx.x;
    if (t >= BS) return;

    float acc[EE];
#pragma unroll
    for (int e = 0; e < EE; ++e) acc[e] = 0.0f;

    const float* xr = x + (size_t)t * HH;
    for (int k = 0; k < HH; ++k) {
        float hv = xr[k];
#pragma unroll
        for (int e = 0; e < EE; ++e) acc[e] += hv * Wr[k * EE + e];
    }

    int best = 0;
    float mx = acc[0];
#pragma unroll
    for (int e = 1; e < EE; ++e) {
        if (acc[e] > mx) { mx = acc[e]; best = e; }   // first-max tie-break
    }
    float s = 0.0f;
#pragma unroll
    for (int e = 0; e < EE; ++e) s += __expf(acc[e] - mx);
    float prob = 1.0f / s;                            // top-1 softmax prob

#pragma unroll
    for (int e = 0; e < EE; ++e) logits_out[(size_t)t * EE + e] = acc[e];
    eidx_out[t] = best;
    prob_ws[t]  = prob;

    int slot = atomicAdd(&counts[best], 1);
    tlist[best * BS + slot] = t;
}

// ---------------------------------------------------------------------------
// Pre-conversion: W1 fp32 [E][H][F] -> bf16 chunk-transposed
//   W1p[((e*FCH + fc)*KCH + kc)*4096 + col*32 + k]
// ---------------------------------------------------------------------------
__global__ __launch_bounds__(256) void moe_conv_w1(
    const float* __restrict__ W1, __bf16* __restrict__ W1p)
{
    int idx = blockIdx.x * 256 + threadIdx.x;   // 0 .. E*F*KCH-1
    int f   = idx % FF;
    int r   = idx / FF;
    int kc  = r % KCH;
    int e   = r / KCH;
    const float* src = W1 + ((size_t)e * HH + kc * 32) * FF + f;
    int fc = f >> 7, col = f & 127;
    __bf16* dst = W1p + (((size_t)(e * FCH + fc) * KCH) + kc) * 4096 + col * 32;
#pragma unroll
    for (int q = 0; q < 8; ++q) {
        v4bf p;
#pragma unroll
        for (int j = 0; j < 4; ++j)
            p[j] = (__bf16)src[(size_t)(q * 4 + j) * FF];
        *(v4bf*)(dst + q * 4) = p;
    }
}

// ---------------------------------------------------------------------------
// Pre-conversion: W2 fp32 [E][F][H] -> bf16 chunk-transposed
//   W2p[(e*K2CH + kc)*24576 + h*32 + k]
// ---------------------------------------------------------------------------
__global__ __launch_bounds__(256) void moe_conv_w2(
    const float* __restrict__ W2, __bf16* __restrict__ W2p)
{
    int idx = blockIdx.x * 256 + threadIdx.x;   // 0 .. E*H*K2CH-1
    int h   = idx % HH;
    int r   = idx / HH;
    int kc  = r % K2CH;
    int e   = r / K2CH;
    const float* src = W2 + ((size_t)e * FF + kc * 32) * HH + h;
    __bf16* dst = W2p + ((size_t)(e * K2CH + kc)) * 24576 + h * 32;
#pragma unroll
    for (int q = 0; q < 8; ++q) {
        v4bf p;
#pragma unroll
        for (int j = 0; j < 4; ++j)
            p[j] = (__bf16)src[(size_t)(q * 4 + j) * HH];
        *(v4bf*)(dst + q * 4) = p;
    }
}

// ---------------------------------------------------------------------------
// FAST kernel: bf16 pre-converted weights, double-buffered async-to-LDS
// staging pipelined against WMMA consumption.
// ---------------------------------------------------------------------------
__global__ __launch_bounds__(256) void moe_expert_mlp_pre(
    const float* __restrict__ x, const __bf16* __restrict__ W1p,
    const __bf16* __restrict__ W2p, const int* __restrict__ counts,
    const int* __restrict__ tlist, const float* __restrict__ prob,
    float* __restrict__ out)
{
    const int e    = blockIdx.x / TILES;
    const int tile = blockIdx.x % TILES;
    const int cnt  = counts[e];
    if (tile * TM >= cnt) return;

    __shared__ __bf16 Xs[32][776];
    __shared__ __bf16 H1s[32][136];
    __shared__ __bf16 W1t[2][128][40];   // double buffer
    __shared__ __bf16 W2t[2][768][40];   // double buffer
    __shared__ int   toks[32];
    __shared__ float probs_s[32];

    const int tid  = threadIdx.x;
    const int lane = tid & 31;
    const int wave = tid >> 5;
    const int m_a  = lane & 15;
    const int hi   = lane >> 4;

    if (tid < TM) {
        int r   = tile * TM + tid;
        int tok = (r < cnt) ? tlist[e * BS + r] : -1;
        toks[tid]    = tok;
        probs_s[tid] = (tok >= 0) ? prob[tok] : 0.0f;
    }
    __syncthreads();

    // Gather X rows -> bf16 LDS (constant trip count: 24*256 == 32*192)
#pragma unroll 1
    for (int it = 0; it < 24; ++it) {
        int i = tid + it * 256;
        int row = i / 192, c4 = i % 192;
        int tok = toks[row];
        float4 v = make_float4(0.f, 0.f, 0.f, 0.f);
        if (tok >= 0) v = ((const float4*)(x + (size_t)tok * HH))[c4];
        v4bf p;
        p[0] = (__bf16)v.x; p[1] = (__bf16)v.y;
        p[2] = (__bf16)v.z; p[3] = (__bf16)v.w;
        *(v4bf*)&Xs[row][c4 * 4] = p;
    }
    __syncthreads();

    // chunk-issue helpers (each thread issues its share of async b128 copies)
    auto issueW1 = [&](int fi, int kc, int d) {
        const __bf16* blk = W1p + (((size_t)(e * FCH + fi) * KCH) + kc) * 4096;
#pragma unroll
        for (int it = 0; it < W1_ITEMS; ++it) {
            int s = tid + it * 256;
            int col = s >> 2, q = s & 3;
            async_cp16(blk + col * 32 + q * 8, &W1t[d][col][q * 8]);
        }
    };
    auto issueW2 = [&](int kc, int d) {
        const __bf16* blk = W2p + ((size_t)(e * K2CH + kc)) * 24576;
#pragma unroll
        for (int it = 0; it < W2_ITEMS; ++it) {
            int s = tid + it * 256;
            int col = s >> 2, q = s & 3;
            async_cp16(blk + col * 32 + q * 8, &W2t[d][col][q * 8]);
        }
    };

    v8f Yacc[2][6];
#pragma unroll
    for (int mt = 0; mt < 2; ++mt)
#pragma unroll
        for (int j = 0; j < 6; ++j)
            Yacc[mt][j] = (v8f){0.f, 0.f, 0.f, 0.f, 0.f, 0.f, 0.f, 0.f};

#pragma unroll 1
    for (int f0 = 0; f0 < FF; f0 += 128) {
        const int fi = f0 >> 7;
        v8f hacc0 = (v8f){0.f, 0.f, 0.f, 0.f, 0.f, 0.f, 0.f, 0.f};
        v8f hacc1 = hacc0;

        // ---- stage 1 (pipelined over KCH k-chunks) ----
        issueW1(fi, 0, 0);
#pragma unroll 1
        for (int c = 0; c < KCH; ++c) {
            if (c + 1 < KCH) {                     // prefetch next chunk
                issueW1(fi, c + 1, (c + 1) & 1);
                async_waitN<W1_ITEMS>();           // chunk c landed (in-order)
            } else {
                async_waitN<0>();
            }
            __syncthreads();

            const int k0 = 32 * c;
            const int d  = c & 1;
            const __bf16* ap0 = &Xs[m_a][k0 + 8 * hi];
            const __bf16* ap1 = &Xs[16 + m_a][k0 + 8 * hi];
            v16bf a0 = CAT16(*(const v8bf*)ap0, *(const v8bf*)(ap0 + 16));
            v16bf a1 = CAT16(*(const v8bf*)ap1, *(const v8bf*)(ap1 + 16));
            const __bf16* bp = &W1t[d][16 * wave + m_a][8 * hi];
            v16bf b = CAT16(*(const v8bf*)bp, *(const v8bf*)(bp + 16));

            hacc0 = __builtin_amdgcn_wmma_f32_16x16x32_bf16(
                false, a0, false, b, (short)0, hacc0, false, false);
            hacc1 = __builtin_amdgcn_wmma_f32_16x16x32_bf16(
                false, a1, false, b, (short)0, hacc1, false, false);
            __syncthreads();
        }

        // overlap first W2 chunk copy with the ReLU/H1 writeout
        issueW2(4 * fi, 0);

#pragma unroll
        for (int r = 0; r < 8; ++r) {
            float t0 = hacc0[r], t1 = hacc1[r];
            H1s[r + 8 * hi][16 * wave + m_a]      = (__bf16)(t0 > 0.f ? t0 : 0.f);
            H1s[16 + r + 8 * hi][16 * wave + m_a] = (__bf16)(t1 > 0.f ? t1 : 0.f);
        }
        __syncthreads();

        // ---- stage 2 (pipelined over 4 k-chunks of W2) ----
#pragma unroll 1
        for (int kk = 0; kk < 4; ++kk) {
            if (kk + 1 < 4) {
                issueW2(4 * fi + kk + 1, (kk + 1) & 1);
                async_waitN<W2_ITEMS>();
            } else {
                async_waitN<0>();
            }
            __syncthreads();

            const int d = kk & 1;
            const __bf16* ap0 = &H1s[m_a][32 * kk + 8 * hi];
            const __bf16* ap1 = &H1s[16 + m_a][32 * kk + 8 * hi];
            v16bf a0 = CAT16(*(const v8bf*)ap0, *(const v8bf*)(ap0 + 16));
            v16bf a1 = CAT16(*(const v8bf*)ap1, *(const v8bf*)(ap1 + 16));

#pragma unroll
            for (int j = 0; j < 6; ++j) {
                const __bf16* bp = &W2t[d][96 * wave + 16 * j + m_a][8 * hi];
                v16bf b2 = CAT16(*(const v8bf*)bp, *(const v8bf*)(bp + 16));
                Yacc[0][j] = __builtin_amdgcn_wmma_f32_16x16x32_bf16(
                    false, a0, false, b2, (short)0, Yacc[0][j], false, false);
                Yacc[1][j] = __builtin_amdgcn_wmma_f32_16x16x32_bf16(
                    false, a1, false, b2, (short)0, Yacc[1][j], false, false);
            }
            __syncthreads();
        }
    }

#pragma unroll
    for (int mt = 0; mt < 2; ++mt) {
#pragma unroll
        for (int r = 0; r < 8; ++r) {
            int m   = 16 * mt + 8 * hi + r;
            int tok = toks[m];
            if (tok >= 0) {
                float ps = probs_s[m];
#pragma unroll
                for (int j = 0; j < 6; ++j) {
                    int col = 96 * wave + 16 * j + m_a;
                    out[(size_t)tok * HH + col] = Yacc[mt][j][r] * ps;
                }
            }
        }
    }
}

// ---------------------------------------------------------------------------
// FALLBACK kernel: fp32 weights staged+converted synchronously.
// ---------------------------------------------------------------------------
__global__ __launch_bounds__(256) void moe_expert_mlp_sync(
    const float* __restrict__ x, const float* __restrict__ W1,
    const float* __restrict__ W2, const int* __restrict__ counts,
    const int* __restrict__ tlist, const float* __restrict__ prob,
    float* __restrict__ out)
{
    const int e    = blockIdx.x / TILES;
    const int tile = blockIdx.x % TILES;
    const int cnt  = counts[e];
    if (tile * TM >= cnt) return;

    __shared__ __bf16 Xs[32][776];
    __shared__ __bf16 H1s[32][136];
    __shared__ __bf16 W1t[128][40];
    __shared__ __bf16 W2t[768][40];
    __shared__ int   toks[32];
    __shared__ float probs_s[32];

    const int tid  = threadIdx.x;
    const int lane = tid & 31;
    const int wave = tid >> 5;
    const int m_a  = lane & 15;
    const int hi   = lane >> 4;

    if (tid < TM) {
        int r   = tile * TM + tid;
        int tok = (r < cnt) ? tlist[e * BS + r] : -1;
        toks[tid]    = tok;
        probs_s[tid] = (tok >= 0) ? prob[tok] : 0.0f;
    }
    __syncthreads();

#pragma unroll 1
    for (int it = 0; it < 24; ++it) {
        int i = tid + it * 256;
        int row = i / 192, c4 = i % 192;
        int tok = toks[row];
        float4 v = make_float4(0.f, 0.f, 0.f, 0.f);
        if (tok >= 0) v = ((const float4*)(x + (size_t)tok * HH))[c4];
        v4bf p;
        p[0] = (__bf16)v.x; p[1] = (__bf16)v.y;
        p[2] = (__bf16)v.z; p[3] = (__bf16)v.w;
        *(v4bf*)&Xs[row][c4 * 4] = p;
    }
    __syncthreads();

    const float* W1e = W1 + (size_t)e * HH * FF;
    const float* W2e = W2 + (size_t)e * FF * HH;

    v8f Yacc[2][6];
#pragma unroll
    for (int mt = 0; mt < 2; ++mt)
#pragma unroll
        for (int j = 0; j < 6; ++j)
            Yacc[mt][j] = (v8f){0.f, 0.f, 0.f, 0.f, 0.f, 0.f, 0.f, 0.f};

#pragma unroll 1
    for (int f0 = 0; f0 < FF; f0 += 128) {
        v8f hacc0 = (v8f){0.f, 0.f, 0.f, 0.f, 0.f, 0.f, 0.f, 0.f};
        v8f hacc1 = hacc0;
#pragma unroll 1
        for (int k0 = 0; k0 < HH; k0 += 32) {
            __syncthreads();
#pragma unroll 1
            for (int it = 0; it < 4; ++it) {       // 4*256 == 128*8
                int i = tid + it * 256;
                int col = i & 127;
                int kq  = i >> 7;
                const float* gp = W1e + (size_t)(k0 + 4 * kq) * FF + f0 + col;
                v4bf p;
#pragma unroll
                for (int j = 0; j < 4; ++j) {
                    p[j] = (__bf16)gp[(size_t)j * FF];
                    __builtin_prefetch(gp + (size_t)j * FF + 32 * FF, 0, 1);
                }
                *(v4bf*)&W1t[col][4 * kq] = p;
            }
            __syncthreads();

            const __bf16* ap0 = &Xs[m_a][k0 + 8 * hi];
            const __bf16* ap1 = &Xs[16 + m_a][k0 + 8 * hi];
            v16bf a0 = CAT16(*(const v8bf*)ap0, *(const v8bf*)(ap0 + 16));
            v16bf a1 = CAT16(*(const v8bf*)ap1, *(const v8bf*)(ap1 + 16));
            const __bf16* bp = &W1t[16 * wave + m_a][8 * hi];
            v16bf b = CAT16(*(const v8bf*)bp, *(const v8bf*)(bp + 16));

            hacc0 = __builtin_amdgcn_wmma_f32_16x16x32_bf16(
                false, a0, false, b, (short)0, hacc0, false, false);
            hacc1 = __builtin_amdgcn_wmma_f32_16x16x32_bf16(
                false, a1, false, b, (short)0, hacc1, false, false);
        }

        __syncthreads();
#pragma unroll
        for (int r = 0; r < 8; ++r) {
            float t0 = hacc0[r], t1 = hacc1[r];
            H1s[r + 8 * hi][16 * wave + m_a]      = (__bf16)(t0 > 0.f ? t0 : 0.f);
            H1s[16 + r + 8 * hi][16 * wave + m_a] = (__bf16)(t1 > 0.f ? t1 : 0.f);
        }
        __syncthreads();

#pragma unroll 1
        for (int kk = 0; kk < 4; ++kk) {
            __syncthreads();
#pragma unroll 1
            for (int it = 0; it < 24; ++it) {      // 24*256 == 768*8
                int i = tid + it * 256;
                int col = i % 768;
                int kq  = i / 768;
                const float* gp = W2e + (size_t)(f0 + 32 * kk + 4 * kq) * HH + col;
                v4bf p;
#pragma unroll
                for (int j = 0; j < 4; ++j) {
                    p[j] = (__bf16)gp[(size_t)j * HH];
                    __builtin_prefetch(gp + (size_t)j * HH + 32 * HH, 0, 1);
                }
                *(v4bf*)&W2t[col][4 * kq] = p;
            }
            __syncthreads();

            const __bf16* ap0 = &H1s[m_a][32 * kk + 8 * hi];
            const __bf16* ap1 = &H1s[16 + m_a][32 * kk + 8 * hi];
            v16bf a0 = CAT16(*(const v8bf*)ap0, *(const v8bf*)(ap0 + 16));
            v16bf a1 = CAT16(*(const v8bf*)ap1, *(const v8bf*)(ap1 + 16));

#pragma unroll
            for (int j = 0; j < 6; ++j) {
                const __bf16* bp = &W2t[96 * wave + 16 * j + m_a][8 * hi];
                v16bf b2 = CAT16(*(const v8bf*)bp, *(const v8bf*)(bp + 16));
                Yacc[0][j] = __builtin_amdgcn_wmma_f32_16x16x32_bf16(
                    false, a0, false, b2, (short)0, Yacc[0][j], false, false);
                Yacc[1][j] = __builtin_amdgcn_wmma_f32_16x16x32_bf16(
                    false, a1, false, b2, (short)0, Yacc[1][j], false, false);
            }
        }
    }

#pragma unroll
    for (int mt = 0; mt < 2; ++mt) {
#pragma unroll
        for (int r = 0; r < 8; ++r) {
            int m   = 16 * mt + 8 * hi + r;
            int tok = toks[m];
            if (tok >= 0) {
                float ps = probs_s[m];
#pragma unroll
                for (int j = 0; j < 6; ++j) {
                    int col = 96 * wave + 16 * j + m_a;
                    out[(size_t)tok * HH + col] = Yacc[mt][j][r] * ps;
                }
            }
        }
    }
}

// ---------------------------------------------------------------------------
// Launch.  d_in = {hidden_states, Wr, W1, W2} fp32.
// d_out = [next_states BSxH f32][router_logits BSxE f32][expert_index BS i32]
// d_ws  = [counts][tlist][prob][(aligned) W1p bf16][W2p bf16]
// ---------------------------------------------------------------------------
extern "C" void kernel_launch(void* const* d_in, const int* in_sizes, int n_in,
                              void* d_out, int out_size, void* d_ws, size_t ws_size,
                              hipStream_t stream) {
    const float* x  = (const float*)d_in[0];
    const float* Wr = (const float*)d_in[1];
    const float* W1 = (const float*)d_in[2];
    const float* W2 = (const float*)d_in[3];

    float* out    = (float*)d_out;
    float* logits = out + (size_t)BS * HH;
    int*   eidx   = (int*)(out + (size_t)BS * HH + (size_t)BS * EE);

    char*  ws     = (char*)d_ws;
    int*   counts = (int*)ws;
    int*   tlist  = counts + EE;
    float* prob   = (float*)(tlist + EE * BS);

    size_t tok_bytes = sizeof(int) * EE + sizeof(int) * EE * BS + sizeof(float) * BS;
    size_t w1p_off   = (tok_bytes + 255) & ~(size_t)255;
    size_t w1p_bytes = (size_t)EE * FCH * KCH * 4096 * sizeof(__bf16);   // 37.75 MB
    size_t w2p_off   = w1p_off + w1p_bytes;
    size_t w2p_bytes = (size_t)EE * K2CH * 24576 * sizeof(__bf16);       // 37.75 MB

    moe_zero_counts<<<1, 32, 0, stream>>>(counts);
    moe_router<<<BS / 256, 256, 0, stream>>>(x, Wr, logits, eidx, prob, counts, tlist);

    if (ws_size >= w2p_off + w2p_bytes) {
        __bf16* W1p = (__bf16*)(ws + w1p_off);
        __bf16* W2p = (__bf16*)(ws + w2p_off);
        moe_conv_w1<<<(EE * FF * KCH) / 256, 256, 0, stream>>>(W1, W1p);
        moe_conv_w2<<<(EE * HH * K2CH) / 256, 256, 0, stream>>>(W2, W2p);
        moe_expert_mlp_pre<<<EE * TILES, 256, 0, stream>>>(
            x, W1p, W2p, counts, tlist, prob, out);
    } else {
        moe_expert_mlp_sync<<<EE * TILES, 256, 0, stream>>>(
            x, W1, W2, counts, tlist, prob, out);
    }
}
